// TemporalGAT_90314572301065
// MI455X (gfx1250) — matrix operand order repaired and checked by reading, not compiled, not run on hardware
//
#include <hip/hip_runtime.h>
#include <hip/hip_bf16.h>
#include <math.h>

// ---------------------------------------------------------------------------
// TemporalGAT forward on gfx1250 (CDNA5).
//  - GEMMs via v_wmma_f32_16x16x32_bf16; each wave computes a 16x128 strip
//    (8 accumulators) so one A-fragment feeds 8 WMMAs.
//  - Packed B weights staged into LDS once per block via the CDNA5 async
//    global->LDS path (ASYNCcnt + s_wait_asynccnt), then read as ds_load_b128.
//  - BatchNorm via two-pass column stats (atomics) + fused affine apply.
//  - GAT edge softmax via 3 edge passes with f32 atomics (the 102MB node
//    feature matrix is L2-resident: 192MB global L2).
//  - GAT output bias is absorbed by the following BatchNorm (per-column
//    constant shifts the mean by itself and cancels) -> skipped.
// ---------------------------------------------------------------------------

typedef __attribute__((ext_vector_type(16))) __bf16 v16bf;
typedef __attribute__((ext_vector_type(8)))  float  v8f;
typedef int v4i __attribute__((ext_vector_type(4)));

#define NCOL 256   // hidden width everywhere after the first linear
#define NHEAD 4

// ---------------------------------------------------------------------- utils

__global__ void k_fill(float* __restrict__ p, float v, long long n) {
  long long i = (long long)blockIdx.x * blockDim.x + threadIdx.x;
  if (i < n) p[i] = v;
}

__device__ inline void atomicMaxF(float* addr, float v) {
  // monotonic two-branch trick: works for mixed-sign floats
  if (v >= 0.0f) atomicMax((int*)addr, __float_as_int(v));
  else           atomicMin((unsigned int*)addr, __float_as_uint(v));
}

// -------------------------------------------------------------- weight packer
// W is [NCOL][K] row-major (out-major, as in the reference's  x @ W.T).
// Bp layout: [K/32][NCOL/16][32 lanes][16 bf16]  -- one contiguous 32B
// fragment per lane per (kb, ntile).  B fragment (32x16 KxN, bf16):
//   lane L: n = L%16, half = L/16, element e -> K = 16*half + e.
__global__ void k_pack_b(const float* __restrict__ W, __bf16* __restrict__ Bp, int K) {
  int tid = blockIdx.x * blockDim.x + threadIdx.x;
  int nt  = (K >> 5) * (NCOL / 16) * 32;
  if (tid >= nt) return;
  int lane = tid & 31;
  int nb   = (tid >> 5) & 15;
  int kb   = tid >> 9;
  int n    = nb * 16 + (lane & 15);
  int half = lane >> 4;
  __bf16* dst = Bp + (long long)tid * 16;
  for (int e = 0; e < 16; ++e) {
    int k = kb * 32 + half * 16 + e;
    dst[e] = (__bf16)W[(long long)n * K + k];
  }
}

// ------------------------------------------------------------------ WMMA GEMM
// C[N,256] = act(A[N,K] @ W.T + bias).
// Block: 128 threads = 4 waves.  Wave w computes rows [m0+w*16, +16) x 128
// cols (col-group blockIdx.x).  All 4 waves share the same 128 columns, so
// the packed-B sub-block ((K/32) x 8 tiles, <=64KB) is staged in LDS once.
// A fragment (16x32 MxK bf16): lane L: m = L%16, half = L/16,
//   element e -> K = 16*half + e
__global__ void __launch_bounds__(128)
k_gemm_wmma(const float* __restrict__ A, const __bf16* __restrict__ Bp,
            float* __restrict__ C, int K, int Nn,
            const float* __restrict__ bias, int relu) {
  __shared__ __bf16 sB[32768];           // 64KB: max (256/32)*8*32*16 bf16

  int wave = threadIdx.x >> 5;
  int lane = threadIdx.x & 31;
  int cg   = blockIdx.x;                 // column group: 128 cols
  int m0   = blockIdx.y * 64 + wave * 16;
  int half = lane >> 4;
  int nk   = K >> 5;

  // ---- stage packed B for this column group into LDS (async DMA path) ----
  // global sub-block per kb: 8 tiles x 32 lanes x 16 bf16 = 4096 elems (8KB)
  for (int kb = 0; kb < nk; ++kb) {
    const __bf16* src = Bp + ((long long)(kb * 16 + cg * 8)) * 32 * 16;
    __bf16*       dst = sB + (long long)kb * 4096;
    for (int off = threadIdx.x * 8; off < 4096; off += 128 * 8) {   // 16B chunks
#if __has_builtin(__builtin_amdgcn_global_load_async_to_lds_b128)
      __builtin_amdgcn_global_load_async_to_lds_b128(
          (__attribute__((address_space(1))) v4i*)(src + off),
          (__attribute__((address_space(3))) v4i*)(dst + off),
          /*imm offset=*/0, /*cpol=*/0);
#else
      *reinterpret_cast<float4*>(dst + off) =
          *reinterpret_cast<const float4*>(src + off);
#endif
    }
  }
#if __has_builtin(__builtin_amdgcn_global_load_async_to_lds_b128)
  asm volatile("s_wait_asynccnt 0" ::: "memory");
#endif
  __syncthreads();

  int mrow = m0 + (lane & 15);
  int mcl  = mrow < Nn ? mrow : (Nn - 1);      // keep EXEC uniform for WMMA
  const float* arow = A + (long long)mcl * K;

  v8f acc[8];
#pragma unroll
  for (int nb = 0; nb < 8; ++nb) acc[nb] = (v8f){};

  for (int kb = 0; kb < nk; ++kb) {
    int kbase = kb * 32;
    const float4* p0 = reinterpret_cast<const float4*>(arow + kbase + 16 * half);
    const float4* p1 = reinterpret_cast<const float4*>(arow + kbase + 16 * half + 8);
    float4 u0 = p0[0], u1 = p0[1];
    float4 w0 = p1[0], w1 = p1[1];
    v16bf af;
    af[0]=(__bf16)u0.x;  af[1]=(__bf16)u0.y;  af[2]=(__bf16)u0.z;  af[3]=(__bf16)u0.w;
    af[4]=(__bf16)u1.x;  af[5]=(__bf16)u1.y;  af[6]=(__bf16)u1.z;  af[7]=(__bf16)u1.w;
    af[8]=(__bf16)w0.x;  af[9]=(__bf16)w0.y;  af[10]=(__bf16)w0.z; af[11]=(__bf16)w0.w;
    af[12]=(__bf16)w1.x; af[13]=(__bf16)w1.y; af[14]=(__bf16)w1.z; af[15]=(__bf16)w1.w;

#pragma unroll
    for (int nb = 0; nb < 8; ++nb) {
      v16bf bf = *reinterpret_cast<const v16bf*>(&sB[((kb * 8 + nb) * 32 + lane) * 16]);
      acc[nb] = __builtin_amdgcn_wmma_f32_16x16x32_bf16(
          /*neg_a=*/false, af, /*neg_b=*/false, bf,
          /*c_mod=*/(short)0, acc[nb], /*reuse_a=*/false, /*reuse_b=*/false);
    }
  }

#pragma unroll
  for (int nb = 0; nb < 8; ++nb) {
    int ncol = cg * 128 + nb * 16 + (lane & 15);
    float bs = bias ? bias[ncol] : 0.0f;
    for (int j = 0; j < 8; ++j) {
      int m = m0 + half * 8 + j;         // C/D layout: lanes 16-31 hold M=j+8
      if (m < Nn) {
        float v = acc[nb][j] + bs;
        if (relu) v = v > 0.0f ? v : 0.0f;
        C[(long long)m * NCOL + ncol] = v;
      }
    }
  }
}

// -------------------------------------------------------------- BN statistics
__global__ void k_col_stats(const float* __restrict__ X, float* __restrict__ sum,
                            float* __restrict__ sumsq, int Nn, int rowsPerBlock) {
  int c  = threadIdx.x;                 // 256 threads = 256 columns
  int r0 = blockIdx.x * rowsPerBlock;
  int r1 = r0 + rowsPerBlock; if (r1 > Nn) r1 = Nn;
  float s = 0.0f, q = 0.0f;
  for (int r = r0; r < r1; ++r) {
    float v = X[(long long)r * NCOL + c];
    s += v; q += v * v;
  }
  atomicAdd(&sum[c], s);
  atomicAdd(&sumsq[c], q);
}

__global__ void k_bn_finalize(const float* __restrict__ sum, const float* __restrict__ sumsq,
                              const float* __restrict__ g, const float* __restrict__ b,
                              float* __restrict__ scale, float* __restrict__ shift, int Nn) {
  int c = threadIdx.x;
  float inv = 1.0f / (float)Nn;
  float mu  = sum[c] * inv;
  float var = sumsq[c] * inv - mu * mu;          // biased variance (jnp.var)
  float sc  = g[c] * rsqrtf(var + 1e-5f);
  scale[c] = sc;
  shift[c] = b[c] - mu * sc;
}

// act: 0 = identity, 1 = relu, 2 = elu
__global__ void k_bn_apply(const float* __restrict__ X, float* __restrict__ Y,
                           const float* __restrict__ scale, const float* __restrict__ shift,
                           long long n, int act) {
  long long i = (long long)blockIdx.x * blockDim.x + threadIdx.x;
  if (i >= n) return;
  int c = (int)(i & (NCOL - 1));
  float v = X[i] * scale[c] + shift[c];
  if (act == 1)      v = v > 0.0f ? v : 0.0f;
  else if (act == 2) v = v > 0.0f ? v : (__expf(v) - 1.0f);
  Y[i] = v;
}

// --------------------------------------------------------- GAT node-side dots
// s[n,h] = sum_c h[n,h,c]*a_src[h,c] ; d likewise.  One wave per node; lane l
// reads channels {j*32+l}, coalesced; per-head partials reduced across wave.
__global__ void k_node_scores(const float* __restrict__ H, const float* __restrict__ a_src,
                              const float* __restrict__ a_dst, float* __restrict__ S,
                              float* __restrict__ D, int Nn) {
  int wave = threadIdx.x >> 5, lane = threadIdx.x & 31;
  int n = blockIdx.x * (blockDim.x >> 5) + wave;
  if (n >= Nn) return;
  const float* hp = H + (long long)n * NCOL;
  float ps[NHEAD] = {0, 0, 0, 0}, pd[NHEAD] = {0, 0, 0, 0};
  for (int j = 0; j < 8; ++j) {
    int c = j * 32 + lane;               // head = c/64 = j>>1
    float hv = hp[c];
    ps[j >> 1] += hv * a_src[c];
    pd[j >> 1] += hv * a_dst[c];
  }
  for (int o = 16; o > 0; o >>= 1)
    for (int h = 0; h < NHEAD; ++h) {
      ps[h] += __shfl_xor(ps[h], o, 32);
      pd[h] += __shfl_xor(pd[h], o, 32);
    }
  if (lane == 0)
    for (int h = 0; h < NHEAD; ++h) {
      S[(long long)n * NHEAD + h] = ps[h];
      D[(long long)n * NHEAD + h] = pd[h];
    }
}

// ------------------------------------------------------------------ edge pass
__device__ inline void edge_ends(const long long* src, const long long* dst,
                                 long long e, long long E, long long& s, long long& d) {
  if (e < E) { s = src[e]; d = dst[e]; }
  else       { s = d = e - E; }          // self loops appended
}

// pass 1: alpha = leaky_relu(s[src]+d[dst]); store; segment max over dst
__global__ void k_edge_alpha_max(const long long* __restrict__ src, const long long* __restrict__ dst,
                                 const float* __restrict__ S, const float* __restrict__ D,
                                 float* __restrict__ ebuf, float* __restrict__ amax,
                                 long long E, long long Etot) {
  long long idx = (long long)blockIdx.x * blockDim.x + threadIdx.x;
  if (idx >= Etot * NHEAD) return;
  long long e = idx >> 2; int h = (int)(idx & 3);
  long long sn, dn; edge_ends(src, dst, e, E, sn, dn);
  float a = S[sn * NHEAD + h] + D[dn * NHEAD + h];
  a = a > 0.0f ? a : 0.2f * a;
  ebuf[idx] = a;
  atomicMaxF(&amax[dn * NHEAD + h], a);
}

// pass 2: e = exp(alpha - amax[dst]); store; segment sum over dst
__global__ void k_edge_expsum(const long long* __restrict__ src, const long long* __restrict__ dst,
                              float* __restrict__ ebuf, const float* __restrict__ amax,
                              float* __restrict__ denom, long long E, long long Etot) {
  long long idx = (long long)blockIdx.x * blockDim.x + threadIdx.x;
  if (idx >= Etot * NHEAD) return;
  long long e = idx >> 2; int h = (int)(idx & 3);
  long long sn, dn; edge_ends(src, dst, e, E, sn, dn);
  float v = __expf(ebuf[idx] - amax[dn * NHEAD + h]);
  ebuf[idx] = v;
  atomicAdd(&denom[dn * NHEAD + h], v);
}

// pass 3: out[dst] += h[src] * (e / denom[dst]).  256 threads = 4 edges/block,
// 64 threads x float4 covering 256 channels; 4 scalar f32 atomics each.
__global__ void k_edge_scatter(const long long* __restrict__ src, const long long* __restrict__ dst,
                               const float* __restrict__ H, const float* __restrict__ ebuf,
                               const float* __restrict__ denom, float* __restrict__ Out,
                               long long E, long long Etot) {
  int le = threadIdx.x >> 6;             // local edge 0..3
  int t  = threadIdx.x & 63;             // float4 slot; head = t/16
  long long e = (long long)blockIdx.x * 4 + le;
  if (e >= Etot) return;
  long long sn, dn; edge_ends(src, dst, e, E, sn, dn);
  int h = t >> 4;
  float w = ebuf[e * NHEAD + h] / (denom[dn * NHEAD + h] + 1e-16f);
  float4 hv = reinterpret_cast<const float4*>(H + sn * NCOL)[t];
  float* op = Out + dn * NCOL + t * 4;
  atomicAdd(op + 0, hv.x * w);
  atomicAdd(op + 1, hv.y * w);
  atomicAdd(op + 2, hv.z * w);
  atomicAdd(op + 3, hv.w * w);
}

// ---------------------------------------------------------------------- driver

static inline long long cdiv(long long a, long long b) { return (a + b - 1) / b; }

extern "C" void kernel_launch(void* const* d_in, const int* in_sizes, int n_in,
                              void* d_out, int out_size, void* d_ws, size_t ws_size,
                              hipStream_t stream) {
  const float*     x      = (const float*)d_in[0];
  const long long* ei     = (const long long*)d_in[1];   // int64 [2,E]
  const float*     w_in1  = (const float*)d_in[2];
  const float*     b_in1  = (const float*)d_in[3];
  const float*     bn_in_g= (const float*)d_in[4];
  const float*     bn_in_b= (const float*)d_in[5];
  const float*     w_in2  = (const float*)d_in[6];
  const float*     b_in2  = (const float*)d_in[7];
  const float*     pn_g   = (const float*)d_in[8];
  const float*     pn_b   = (const float*)d_in[9];
  const float*     gat1_W = (const float*)d_in[10];
  const float*     gat1_as= (const float*)d_in[11];
  const float*     gat1_ad= (const float*)d_in[12];
  // d_in[13] = gat1_b : absorbed by bn1 (per-column pre-BN bias cancels)
  const float*     bn1_g  = (const float*)d_in[14];
  const float*     bn1_b  = (const float*)d_in[15];
  const float*     gat2_W = (const float*)d_in[16];
  const float*     gat2_as= (const float*)d_in[17];
  const float*     gat2_ad= (const float*)d_in[18];
  // d_in[19] = gat2_b : absorbed by bn2
  const float*     bn2_g  = (const float*)d_in[20];
  const float*     bn2_b  = (const float*)d_in[21];

  const int  F_IN = 128;
  const int  Nn   = in_sizes[0] / F_IN;
  const long long E    = (long long)in_sizes[1] / 2;
  const long long Etot = E + Nn;
  const long long* esrc = ei;
  const long long* edst = ei + E;

  // ---- workspace carve (256B aligned slices) ----
  char* p = (char*)d_ws;
  auto take = [&](size_t bytes) { char* r = p; p += (bytes + 255) & ~(size_t)255; return (void*)r; };
  float*  bufA  = (float*)take((size_t)Nn * NCOL * 4);
  float*  bufB  = (float*)take((size_t)Nn * NCOL * 4);
  __bf16* pW1   = (__bf16*)take((size_t)(F_IN / 32) * 16 * 32 * 16 * 2);
  __bf16* pW2   = (__bf16*)take((size_t)(NCOL / 32) * 16 * 32 * 16 * 2);
  __bf16* pG1   = (__bf16*)take((size_t)(NCOL / 32) * 16 * 32 * 16 * 2);
  __bf16* pG2   = (__bf16*)take((size_t)(NCOL / 32) * 16 * 32 * 16 * 2);
  float*  Sv    = (float*)take((size_t)Nn * NHEAD * 4);
  float*  Dv    = (float*)take((size_t)Nn * NHEAD * 4);
  float*  amax  = (float*)take((size_t)Nn * NHEAD * 4);
  float*  denom = (float*)take((size_t)Nn * NHEAD * 4);
  float*  ebuf  = (float*)take((size_t)Etot * NHEAD * 4);
  float*  stSum = (float*)take(256 * 4);
  float*  stSq  = (float*)take(256 * 4);
  float*  stScl = (float*)take(256 * 4);
  float*  stSft = (float*)take(256 * 4);
  float*  out   = (float*)d_out;

  const long long NC = (long long)Nn * NCOL;
  const int ROWSPB = 512;
  dim3 gGemm(2, (unsigned)cdiv(Nn, 64));   // 2 col-groups x row blocks of 64
  dim3 bGemm(128);
  long long eh = Etot * NHEAD;

  auto gemm = [&](const float* A, const __bf16* Bp, float* C, int K,
                  const float* bias, int relu) {
    k_gemm_wmma<<<gGemm, bGemm, 0, stream>>>(A, Bp, C, K, Nn, bias, relu);
  };
  auto bn = [&](const float* X, float* Y, const float* g, const float* b, int act) {
    k_fill<<<(unsigned)cdiv(512, 256), 256, 0, stream>>>(stSum, 0.0f, 512); // sum+sq contiguous
    k_col_stats<<<(unsigned)cdiv(Nn, ROWSPB), 256, 0, stream>>>(X, stSum, stSq, Nn, ROWSPB);
    k_bn_finalize<<<1, 256, 0, stream>>>(stSum, stSq, g, b, stScl, stSft, Nn);
    k_bn_apply<<<(unsigned)cdiv(NC, 256), 256, 0, stream>>>(X, Y, stScl, stSft, NC, act);
  };
  auto gat = [&](const float* H, const float* as, const float* ad, float* Out) {
    k_node_scores<<<(unsigned)cdiv(Nn, 8), 256, 0, stream>>>(H, as, ad, Sv, Dv, Nn);
    k_fill<<<(unsigned)cdiv((long long)Nn * NHEAD, 256), 256, 0, stream>>>(amax, -1e30f, (long long)Nn * NHEAD);
    k_fill<<<(unsigned)cdiv((long long)Nn * NHEAD, 256), 256, 0, stream>>>(denom, 0.0f, (long long)Nn * NHEAD);
    k_fill<<<(unsigned)cdiv(NC, 256), 256, 0, stream>>>(Out, 0.0f, NC);
    k_edge_alpha_max<<<(unsigned)cdiv(eh, 256), 256, 0, stream>>>(esrc, edst, Sv, Dv, ebuf, amax, E, Etot);
    k_edge_expsum   <<<(unsigned)cdiv(eh, 256), 256, 0, stream>>>(esrc, edst, ebuf, amax, denom, E, Etot);
    k_edge_scatter  <<<(unsigned)cdiv(Etot, 4), 256, 0, stream>>>(esrc, edst, H, ebuf, denom, Out, E, Etot);
  };

  // ---- pack weights into WMMA B-fragment layout ----
  k_pack_b<<<(unsigned)cdiv((F_IN / 32) * 16 * 32, 256), 256, 0, stream>>>(w_in1, pW1, F_IN);
  k_pack_b<<<(unsigned)cdiv((NCOL / 32) * 16 * 32, 256), 256, 0, stream>>>(w_in2, pW2, NCOL);
  k_pack_b<<<(unsigned)cdiv((NCOL / 32) * 16 * 32, 256), 256, 0, stream>>>(gat1_W, pG1, NCOL);
  k_pack_b<<<(unsigned)cdiv((NCOL / 32) * 16 * 32, 256), 256, 0, stream>>>(gat2_W, pG2, NCOL);

  // ---- input projection: Linear+ReLU -> BN -> Linear -> BN+ReLU ----
  gemm(x, pW1, bufA, F_IN, b_in1, /*relu=*/1);
  bn(bufA, out, bn_in_g, bn_in_b, /*act=*/0);          // d_out as scratch
  gemm(out, pW2, bufB, NCOL, b_in2, /*relu=*/0);
  bn(bufB, bufA, pn_g, pn_b, /*act=*/1);

  // ---- GAT layer 1 ----
  gemm(bufA, pG1, bufB, NCOL, nullptr, 0);             // h = x @ W.T
  gat(bufB, gat1_as, gat1_ad, out);                    // aggregate into d_out
  bn(out, bufA, bn1_g, bn1_b, /*act=*/2);              // ELU(BN(.)) (bias absorbed)

  // ---- GAT layer 2 ----
  gemm(bufA, pG2, bufB, NCOL, nullptr, 0);
  gat(bufB, gat2_as, gat2_ad, bufA);                   // aggregate into bufA
  bn(bufA, out, bn2_g, bn2_b, /*act=*/2);              // final output
}
